// FlowMatchingDurationHead_8040178778621
// MI455X (gfx1250) — compile-verified, matrix operand order checked by reading, not run on hardware
//
#include <hip/hip_runtime.h>
#include <hip/hip_bf16.h>
#include <math.h>

// ---------------------------------------------------------------------------
// FlowMatchingDurationHead forward for MI455X (gfx1250, wave32, WMMA bf16,
// TDM async tensor staging for the GEMMs).
// ---------------------------------------------------------------------------

typedef __attribute__((ext_vector_type(16))) __bf16 v16bf;
typedef __attribute__((ext_vector_type(8)))  float  v8f;
typedef __attribute__((ext_vector_type(4)))  unsigned int v4u;
typedef __attribute__((ext_vector_type(8)))  int  v8i;
typedef __attribute__((ext_vector_type(4)))  int  v4i;

#define Bsz 8
#define Tn  1024
#define Dm  512
#define Hn  8
#define HDm 64
#define FFd 2048
#define Lnum 4
#define Mrows (Bsz*Tn)   /* 8192 */

#define WMMA_BF16(a,b,c) \
  __builtin_amdgcn_wmma_f32_16x16x32_bf16(false,(a),false,(b),(short)0,(c),false,false)

__device__ __forceinline__ unsigned short f2bf(float x) {
  unsigned int u = __float_as_uint(x);
  u += 0x7fffu + ((u >> 16) & 1u);          // round-to-nearest-even
  return (unsigned short)(u >> 16);
}

// Load a 16x32 bf16 WMMA fragment (A or B operand). ISA layout (CDNA5 ISA):
// lane&15 selects the row (M for A / N for B); lane>>4 selects which 8-wide
// K-chunk pair: K-chunks at k0+(lane>>4)*8 and k0+16+(lane>>4)*8.
__device__ __forceinline__ v16bf frag_ld(const unsigned short* base, int stride,
                                         int rowbase, int k0, int lane) {
  const unsigned short* p =
      base + (size_t)(rowbase + (lane & 15)) * stride + k0 + ((lane >> 4) << 3);
  union { uint4 q[2]; v16bf v; } u;
  u.q[0] = *(const uint4*)p;
  u.q[1] = *(const uint4*)(p + 16);
  return u.v;
}

// ---------------------------------------------------------------------------
// Tensor Data Mover: issue a 2D tile load (rows x 64 bf16 elements, row-major
// compacted into LDS) described by a D# built per CDNA5 ISA
// §8. Group0: count=1 | lds_addr | global_addr | type=2. Group1: data_size=2B,
// tensor_dim0=row length, tile_dim0=64, tile_dim1=rows, dim0_stride=row stride.
// Tracked by TENSORcnt.
// ---------------------------------------------------------------------------
__device__ __forceinline__ void tdm_load_2d(const unsigned short* gsrc,
                                            unsigned lds_byte_addr,
                                            unsigned rows,
                                            unsigned row_stride_elems) {
  unsigned long long ga = (unsigned long long)(size_t)gsrc;
  unsigned td0 = row_stride_elems;          // tensor dim0 (elements)
  unsigned td1 = 1u << 20;                  // generous dim1 (tiles always fit)

  v4u g0;
  g0[0] = 1u;                                             // count=1, user mode
  g0[1] = lds_byte_addr;                                  // lds_addr [63:32]
  g0[2] = (unsigned)(ga & 0xffffffffu);                   // global_addr lo
  g0[3] = (unsigned)((ga >> 32) & 0x01ffffffu) | (2u << 30); // addr hi | type=2

  v8i g1;
  g1[0] = (int)(1u << 16);                                // data_size=1 (2B)
  g1[1] = (int)((td0 & 0xffffu) << 16);                   // tensor_dim0 lo16
  g1[2] = (int)((td0 >> 16) | ((td1 & 0xffffu) << 16));   // dim0 hi | dim1 lo
  g1[3] = (int)((td1 >> 16) | (64u << 16));               // dim1 hi | tile_dim0=64
  g1[4] = (int)rows;                                      // tile_dim1 | tile_dim2=0
  g1[5] = (int)row_stride_elems;                          // dim0_stride lo32
  g1[6] = 0;                                              // stride hi | dim1_stride lo
  g1[7] = 0;

  v4i g2 = {0, 0, 0, 0};
  v4i g3 = {0, 0, 0, 0};
#if defined(__clang_major__) && (__clang_major__ >= 23)
  v8i g4 = {0, 0, 0, 0, 0, 0, 0, 0};
  __builtin_amdgcn_tensor_load_to_lds(g0, g1, g2, g3, g4, 0);
#else
  __builtin_amdgcn_tensor_load_to_lds(g0, g1, g2, g3, 0);
#endif
}

// ---------------------------------------------------------------------------
// fp32 -> bf16 weight conversion
// ---------------------------------------------------------------------------
__global__ void __launch_bounds__(256) cvt_kernel(const float* __restrict__ src,
                                                  unsigned short* __restrict__ dst,
                                                  int n) {
  int i = blockIdx.x * 256 + threadIdx.x;
  if (i < n) dst[i] = f2bf(src[i]);
}

// ---------------------------------------------------------------------------
// timestep embedding + t_proj:  temb[b, :] = sincos(t_b) @ t_proj_w^T + t_proj_b
// ---------------------------------------------------------------------------
__global__ void __launch_bounds__(256) temb_kernel(const float* __restrict__ t,
                                                   const float* __restrict__ tw,
                                                   const float* __restrict__ tb,
                                                   float* __restrict__ temb) {
  __shared__ float emb[Dm];
  int b = blockIdx.x, tid = threadIdx.x;
  float tv = t[b];
  if (tid < 256) {
    float f = __expf(-0.035977892f * (float)tid);  // exp(-ln(10000)*k/256)
    float a = tv * f;
    emb[tid]       = sinf(a);
    emb[tid + 256] = cosf(a);
  }
  __syncthreads();
  for (int d = tid; d < Dm; d += 256) {
    float acc = tb[d];
    const float* wrow = tw + (size_t)d * Dm;
    for (int k = 0; k < Dm; ++k) acc += emb[k] * wrow[k];
    temb[b * Dm + d] = acc;
  }
}

// ---------------------------------------------------------------------------
// z = h + x_t * x_proj_w + x_proj_b + temb[b]
// ---------------------------------------------------------------------------
__global__ void __launch_bounds__(256) prep_kernel(const float* __restrict__ h,
                                                   const float* __restrict__ x_t,
                                                   const float* __restrict__ xw,
                                                   const float* __restrict__ xb,
                                                   const float* __restrict__ temb,
                                                   float* __restrict__ z) {
  size_t i = (size_t)blockIdx.x * 256 + threadIdx.x;
  if (i >= (size_t)Mrows * Dm) return;
  int d = (int)(i & (Dm - 1));
  size_t tok = i >> 9;
  int b = (int)(tok >> 10);
  z[i] = h[i] + x_t[tok] * xw[d] + xb[d] + temb[b * Dm + d];
}

// ---------------------------------------------------------------------------
// LayerNorm over D=512 per token, bf16 output. One wave (32 lanes) per token.
// ---------------------------------------------------------------------------
__global__ void __launch_bounds__(256) ln_kernel(const float* __restrict__ z,
                                                 const float* __restrict__ w,
                                                 const float* __restrict__ bb,
                                                 unsigned short* __restrict__ u) {
  int token = blockIdx.x * 8 + (threadIdx.x >> 5);
  int lane = threadIdx.x & 31;
  const float* x = z + (size_t)token * Dm;
  float v[16];
  float s = 0.f;
#pragma unroll
  for (int i = 0; i < 16; ++i) { v[i] = x[lane + i * 32]; s += v[i]; }
#pragma unroll
  for (int o = 16; o >= 1; o >>= 1) s += __shfl_xor(s, o, 32);
  float mu = s * (1.0f / 512.0f);
  float var = 0.f;
#pragma unroll
  for (int i = 0; i < 16; ++i) { float d = v[i] - mu; var += d * d; }
#pragma unroll
  for (int o = 16; o >= 1; o >>= 1) var += __shfl_xor(var, o, 32);
  float inv = rsqrtf(var * (1.0f / 512.0f) + 1e-5f);
#pragma unroll
  for (int i = 0; i < 16; ++i) {
    int d = lane + i * 32;
    u[(size_t)token * Dm + d] = f2bf((v[i] - mu) * inv * w[d] + bb[d]);
  }
}

// ---------------------------------------------------------------------------
// GEMM: C[M,N] = A[M,K] @ W[N,K]^T + bias   (A,W bf16, accumulate f32)
// block = 256 threads (8 waves); tile 128(M) x 64(N); K-step 64.
// Staging: TDM async tensor loads into double-buffered LDS (wave 0 issues the
// descriptors; completion via s_wait_tensorcnt; visibility via barrier).
// modes: 0 = bf16 out | 1 = QKV (scale first 512 cols by 1/8, bf16 out)
//        2 = residual fp32 add into zres | 3 = exact GELU, bf16 out
// ---------------------------------------------------------------------------
__global__ void __launch_bounds__(256) gemm_kernel(const unsigned short* __restrict__ A,
                                                   const unsigned short* __restrict__ W,
                                                   const float* __restrict__ bias,
                                                   unsigned short* __restrict__ outb,
                                                   float* __restrict__ zres,
                                                   int K, int N, int mode) {
  __shared__ __align__(16) unsigned short sA[2][128 * 64];   // 2 x 16KB
  __shared__ __align__(16) unsigned short sB[2][64 * 64];    // 2 x  8KB
  int tid = threadIdx.x, lane = tid & 31, w = tid >> 5;
  int wm = w & 3, wn = w >> 2;            // 4 x 2 wave grid, 32x32 per wave
  int m0 = blockIdx.x * 128, n0 = blockIdx.y * 64;
  v8f acc[2][2] = {};

  unsigned sAaddr = (unsigned)(unsigned long long)(size_t)&sA[0][0];
  unsigned sBaddr = (unsigned)(unsigned long long)(size_t)&sB[0][0];
  int nk = K >> 6;                        // K-steps of 64

  if (w == 0) {                           // prologue: DMA tile 0 into buffer 0
    tdm_load_2d(A + (size_t)m0 * K, sAaddr, 128, (unsigned)K);
    tdm_load_2d(W + (size_t)n0 * K, sBaddr, 64, (unsigned)K);
  }

  for (int i = 0; i < nk; ++i) {
    int cur = i & 1, nxt = cur ^ 1;
    if (w == 0) {
      if (i + 1 < nk) {                   // prefetch next tile into other buffer
        tdm_load_2d(A + (size_t)m0 * K + (size_t)(i + 1) * 64,
                    sAaddr + (unsigned)nxt * (128 * 64 * 2), 128, (unsigned)K);
        tdm_load_2d(W + (size_t)n0 * K + (size_t)(i + 1) * 64,
                    sBaddr + (unsigned)nxt * (64 * 64 * 2), 64, (unsigned)K);
        __builtin_amdgcn_s_wait_tensorcnt((short)2);  // current pair landed
      } else {
        __builtin_amdgcn_s_wait_tensorcnt((short)0);
      }
    }
    __syncthreads();                      // TDM data visible to all waves

#pragma unroll
    for (int s = 0; s < 2; ++s) {         // two 32-deep K sub-steps
      v16bf a0 = frag_ld(&sA[cur][0], 64, wm * 32,      s * 32, lane);
      v16bf a1 = frag_ld(&sA[cur][0], 64, wm * 32 + 16, s * 32, lane);
      v16bf b0 = frag_ld(&sB[cur][0], 64, wn * 32,      s * 32, lane);
      v16bf b1 = frag_ld(&sB[cur][0], 64, wn * 32 + 16, s * 32, lane);
      acc[0][0] = WMMA_BF16(a0, b0, acc[0][0]);
      acc[0][1] = WMMA_BF16(a0, b1, acc[0][1]);
      acc[1][0] = WMMA_BF16(a1, b0, acc[1][0]);
      acc[1][1] = WMMA_BF16(a1, b1, acc[1][1]);
    }
    __syncthreads();                      // reads done before buffer reuse
  }

  // epilogue: C/D layout = row e+8*(lane>>4), col lane&15
#pragma unroll
  for (int mi = 0; mi < 2; ++mi)
#pragma unroll
    for (int ni = 0; ni < 2; ++ni) {
      int gcol = n0 + wn * 32 + ni * 16 + (lane & 15);
      float bv = bias[gcol];
#pragma unroll
      for (int e = 0; e < 8; ++e) {
        int grow = m0 + wm * 32 + mi * 16 + e + ((lane >> 4) << 3);
        float c = acc[mi][ni][e] + bv;
        if (mode == 1 && gcol < Dm) c *= 0.125f;                   // q * 1/sqrt(hd)
        if (mode == 3) c = 0.5f * c * (1.0f + erff(c * 0.70710678118f));
        if (mode == 2) zres[(size_t)grow * Dm + gcol] += c;
        else           outb[(size_t)grow * N + gcol] = f2bf(c);
      }
    }
}

// ---------------------------------------------------------------------------
// Banded local attention (|i-j| <= 32) per (b, h, 64-query tile).
// 128 threads = 4 waves; each wave owns 16 queries (80-key band, padded to 96).
// S = Q K^T via WMMA; softmax through LDS; O = P V via WMMA (V^T staged).
// qkv layout: bf16 [B*T, 1536] with q|k|v slabs; q pre-scaled by 1/8.
// ---------------------------------------------------------------------------
__global__ void __launch_bounds__(128) attn_kernel(const unsigned short* __restrict__ qkv,
                                                   const unsigned char* __restrict__ pad,
                                                   unsigned short* __restrict__ obuf) {
  __shared__ __align__(16) unsigned short sK[128 * 64];    // [local key][d]
  __shared__ __align__(16) unsigned short sVT[64 * 160];   // [d][local key], padded
  __shared__ __align__(16) float sSP[4 * 16 * 80];         // per-wave S, P overlaid

  int tid = threadIdx.x, lane = tid & 31, w = tid >> 5;
  int qt = blockIdx.x, h = blockIdx.y, b = blockIdx.z;
  int t0 = qt * 64;                       // first query of block; keys t0-32..t0+95

  // stage K tile (zero-padded outside [0,T))
#pragma unroll
  for (int it = 0; it < 8; ++it) {
    int task = it * 128 + tid;
    int j = task >> 3, ch = (task & 7) << 3;
    int g = t0 - 32 + j;
    uint4 val = make_uint4(0u, 0u, 0u, 0u);
    if (g >= 0 && g < Tn)
      val = *(const uint4*)(qkv + (size_t)(b * Tn + g) * 1536 + Dm + h * HDm + ch);
    *(uint4*)(sK + j * 64 + ch) = val;
  }
  // stage V^T tile (transpose on the fly; 64 d-rows x 160 keys)
  for (int it = 0; it < 10; ++it) {
    int task = it * 128 + tid;
    int d = task / 20, kc = task % 20;
    union { unsigned short s[8]; uint4 q; } tmp;
#pragma unroll
    for (int i2 = 0; i2 < 8; ++i2) {
      int g = t0 - 32 + kc * 8 + i2;
      tmp.s[i2] = (g >= 0 && g < Tn)
          ? qkv[(size_t)(b * Tn + g) * 1536 + 2 * Dm + h * HDm + d]
          : (unsigned short)0;
    }
    *(uint4*)(sVT + d * 160 + kc * 8) = tmp.q;
  }
  __syncthreads();

  // Q fragments straight from global (row = query, contiguous head-dim)
  const unsigned short* qbase = qkv + (size_t)(b * Tn + t0 + w * 16) * 1536 + h * HDm;
  v16bf aq0 = frag_ld(qbase, 1536, 0, 0,  lane);
  v16bf aq1 = frag_ld(qbase, 1536, 0, 32, lane);

  float* sw = sSP + w * 1280;             // wave scratch: 16 rows x 80 f32

  // scores: 5 key tiles of 16, K-dim 64 -> 2 WMMA each
#pragma unroll
  for (int kt = 0; kt < 5; ++kt) {
    v16bf bk0 = frag_ld(sK, 64, (w + kt) * 16, 0,  lane);
    v16bf bk1 = frag_ld(sK, 64, (w + kt) * 16, 32, lane);
    v8f s = {};
    s = WMMA_BF16(aq0, bk0, s);
    s = WMMA_BF16(aq1, bk1, s);
#pragma unroll
    for (int e = 0; e < 8; ++e)
      sw[(e + ((lane >> 4) << 3)) * 80 + kt * 16 + (lane & 15)] = s[e];
  }
  __syncthreads();

  // softmax: lane r (<16) owns query row r; band => cols c in [r, r+64]
  if (lane < 16) {
    int r = lane;
    float m = -1e30f;
    for (int c = r; c <= r + 64; ++c) {
      int g = t0 - 32 + w * 16 + c;
      if (g < 0 || g >= Tn || pad[b * Tn + g]) continue;
      m = fmaxf(m, sw[r * 80 + c]);
    }
    float sum = 0.f;
    for (int c = r; c <= r + 64; ++c) {
      int g = t0 - 32 + w * 16 + c;
      if (g < 0 || g >= Tn || pad[b * Tn + g]) continue;
      sum += __expf(sw[r * 80 + c] - m);
    }
    float inv = 1.0f / sum;
    unsigned short* pw16 = (unsigned short*)sw;   // P overlays S (trailing writes)
    for (int c = 0; c < 96; ++c) {
      float p = 0.f;
      if (c >= r && c <= r + 64 && c < 80) {
        int g = t0 - 32 + w * 16 + c;
        if (g >= 0 && g < Tn && !pad[b * Tn + g])
          p = __expf(sw[r * 80 + c] - m) * inv;
      }
      pw16[r * 160 + c] = f2bf(p);                // row stride 320B
    }
  }
  __syncthreads();

  // O = P @ V : 4 d-tiles x 3 K-steps (96 padded keys)
  const unsigned short* pw = (const unsigned short*)sw;
#pragma unroll
  for (int n = 0; n < 4; ++n) {
    v8f o = {};
#pragma unroll
    for (int kt = 0; kt < 3; ++kt) {
      v16bf ap = frag_ld(pw, 160, 0, kt * 32, lane);
      v16bf bv = frag_ld(sVT, 160, n * 16, w * 16 + kt * 32, lane);
      o = WMMA_BF16(ap, bv, o);
    }
#pragma unroll
    for (int e = 0; e < 8; ++e) {
      int t = t0 + w * 16 + e + ((lane >> 4) << 3);
      obuf[(size_t)(b * Tn + t) * Dm + h * HDm + n * 16 + (lane & 15)] = f2bf(o[e]);
    }
  }
}

// ---------------------------------------------------------------------------
// head: out[token] = z[token,:] . head_w + head_b   (one wave per token)
// ---------------------------------------------------------------------------
__global__ void __launch_bounds__(256) head_kernel(const float* __restrict__ z,
                                                   const float* __restrict__ hw,
                                                   const float* __restrict__ hb,
                                                   float* __restrict__ out) {
  int token = blockIdx.x * 8 + (threadIdx.x >> 5);
  int lane = threadIdx.x & 31;
  float s = 0.f;
#pragma unroll
  for (int i = 0; i < 16; ++i) {
    int d = lane + i * 32;
    s += z[(size_t)token * Dm + d] * hw[d];
  }
#pragma unroll
  for (int o = 16; o >= 1; o >>= 1) s += __shfl_xor(s, o, 32);
  if (lane == 0) out[token] = s + hb[0];
}

// ---------------------------------------------------------------------------
extern "C" void kernel_launch(void* const* d_in, const int* in_sizes, int n_in,
                              void* d_out, int out_size, void* d_ws, size_t ws_size,
                              hipStream_t stream) {
  const float* h    = (const float*)d_in[0];
  const float* x_t  = (const float*)d_in[1];
  const float* t    = (const float*)d_in[2];
  const unsigned char* mask = (const unsigned char*)d_in[3];
  const float* xw   = (const float*)d_in[4];
  const float* xb   = (const float*)d_in[5];
  const float* tw   = (const float*)d_in[6];
  const float* tb   = (const float*)d_in[7];
  const float* qkvW = (const float*)d_in[8];
  const float* qkvB = (const float*)d_in[9];
  const float* aoW  = (const float*)d_in[10];
  const float* aoB  = (const float*)d_in[11];
  const float* ln1w = (const float*)d_in[12];
  const float* ln1b = (const float*)d_in[13];
  const float* ln2w = (const float*)d_in[14];
  const float* ln2b = (const float*)d_in[15];
  const float* f1W  = (const float*)d_in[16];
  const float* f1B  = (const float*)d_in[17];
  const float* f2W  = (const float*)d_in[18];
  const float* f2B  = (const float*)d_in[19];
  const float* hw   = (const float*)d_in[20];
  const float* hb   = (const float*)d_in[21];
  float* out = (float*)d_out;

  char* ws = (char*)d_ws;
  size_t off = 0;
  auto alloc = [&](size_t bytes) -> void* {
    void* p = ws + off;
    off = (off + bytes + 255) & ~(size_t)255;
    return p;
  };
  unsigned short* wq  = (unsigned short*)alloc((size_t)Lnum * 1536 * 512 * 2);
  unsigned short* wo  = (unsigned short*)alloc((size_t)Lnum * 512 * 512 * 2);
  unsigned short* w1  = (unsigned short*)alloc((size_t)Lnum * 2048 * 512 * 2);
  unsigned short* w2  = (unsigned short*)alloc((size_t)Lnum * 512 * 2048 * 2);
  float*          z   = (float*)alloc((size_t)Mrows * Dm * 4);
  float*          te  = (float*)alloc((size_t)Bsz * Dm * 4);
  unsigned short* u   = (unsigned short*)alloc((size_t)Mrows * Dm * 2);
  unsigned short* qkvb= (unsigned short*)alloc((size_t)Mrows * 1536 * 2);
  unsigned short* ob  = (unsigned short*)alloc((size_t)Mrows * Dm * 2);
  unsigned short* ffb = (unsigned short*)alloc((size_t)Mrows * FFd * 2);

  // one-time (per launch) weight conversions fp32 -> bf16
  int nq = Lnum * 1536 * 512, no = Lnum * 512 * 512,
      n1 = Lnum * 2048 * 512, n2 = Lnum * 512 * 2048;
  cvt_kernel<<<(nq + 255) / 256, 256, 0, stream>>>(qkvW, wq, nq);
  cvt_kernel<<<(no + 255) / 256, 256, 0, stream>>>(aoW,  wo, no);
  cvt_kernel<<<(n1 + 255) / 256, 256, 0, stream>>>(f1W,  w1, n1);
  cvt_kernel<<<(n2 + 255) / 256, 256, 0, stream>>>(f2W,  w2, n2);

  // input fuse
  temb_kernel<<<Bsz, 256, 0, stream>>>(t, tw, tb, te);
  prep_kernel<<<(Mrows * Dm + 255) / 256, 256, 0, stream>>>(h, x_t, xw, xb, te, z);

  for (int l = 0; l < Lnum; ++l) {
    // pre-norm attention
    ln_kernel<<<Mrows / 8, 256, 0, stream>>>(z, ln1w + l * Dm, ln1b + l * Dm, u);
    gemm_kernel<<<dim3(Mrows / 128, 1536 / 64), 256, 0, stream>>>(
        u, wq + (size_t)l * 1536 * 512, qkvB + l * 1536, qkvb, nullptr, 512, 1536, 1);
    attn_kernel<<<dim3(Tn / 64, Hn, Bsz), 128, 0, stream>>>(qkvb, mask, ob);
    gemm_kernel<<<dim3(Mrows / 128, 512 / 64), 256, 0, stream>>>(
        ob, wo + (size_t)l * 512 * 512, aoB + l * 512, nullptr, z, 512, 512, 2);
    // pre-norm FFN
    ln_kernel<<<Mrows / 8, 256, 0, stream>>>(z, ln2w + l * Dm, ln2b + l * Dm, u);
    gemm_kernel<<<dim3(Mrows / 128, 2048 / 64), 256, 0, stream>>>(
        u, w1 + (size_t)l * 2048 * 512, f1B + l * 2048, ffb, nullptr, 512, 2048, 3);
    gemm_kernel<<<dim3(Mrows / 128, 512 / 64), 256, 0, stream>>>(
        ffb, w2 + (size_t)l * 512 * 2048, f2B + l * 512, nullptr, z, 2048, 512, 2);
  }

  head_kernel<<<Mrows / 8, 256, 0, stream>>>(z, hw, hb, out);
  (void)in_sizes; (void)n_in; (void)out_size; (void)ws_size;
}
